// KSpaceTransformerGNNEncoder_5806795784729
// MI455X (gfx1250) — matrix operand order repaired and checked by reading, not compiled, not run on hardware
//
#include <hip/hip_runtime.h>
#include <hip/hip_bf16.h>
#include <math.h>

#define NODES  20000
#define EDGES  160000
#define HIDDIM 128
#define NHEADS 8
#define NGRAPH 64
// bf16 elements per packed 16-col weight tile: 4 k-chunks * 32 lanes * 16
#define TILE_ELEMS (4 * 32 * 16)

typedef __attribute__((ext_vector_type(16))) __bf16 v16bf;
typedef __attribute__((ext_vector_type(8)))  float  v8f;

__device__ __forceinline__ __bf16 f2bf(float f) {
    unsigned u = __float_as_uint(f);
    u += 0x7FFFu + ((u >> 16) & 1u);          // round-to-nearest-even
    unsigned short h = (unsigned short)(u >> 16);
    return __builtin_bit_cast(__bf16, h);
}

// ---------------------------------------------------------------------------
// Pack a [K=128 x N] f32 weight matrix (row stride ldb) into bf16 fragments
// laid out so the GEMM loads one contiguous 32B vector per lane per k-chunk:
//   pack[((nt*4 + kc)*32 + lane)*16 + s] = bf16( W[(kc*32 + kh + (s&7) + (s>>3)*16)*ldb + nt*16 + (lane&15)] )
// where kh = (lane>=16) ? 8 : 0  (ISA B-matrix 32x16 16-bit layout).
// ---------------------------------------------------------------------------
__global__ void pack_b_kernel(const float* __restrict__ W, int ldb,
                              __bf16* __restrict__ pack, int ntiles)
{
    int idx = blockIdx.x * blockDim.x + threadIdx.x;
    int total = ntiles * TILE_ELEMS;
    if (idx >= total) return;
    int s    = idx & 15;
    int lane = (idx >> 4) & 31;
    int kc   = (idx >> 9) & 3;
    int nt   = idx >> 11;
    int n = nt * 16 + (lane & 15);
    int k = kc * 32 + ((lane >> 4) << 3) + (s & 7) + ((s >> 3) << 4);
    pack[idx] = f2bf(W[(size_t)k * ldb + n]);
}

// ---------------------------------------------------------------------------
// WMMA GEMM:  C[M x 128] = A[M x 128] * Bpack + bias.   REQUIRES M % 16 == 0.
// Fixed strides (128) so B loads and C stores use immediate offsets.
// One wave per 16-row tile. A converted to bf16 once per wave (float4 loads).
// Per column tile: preload 4 B fragments (clause), then 4 chained WMMAs.
// ---------------------------------------------------------------------------
__global__ __launch_bounds__(128)
void wmma_gemm_bias(const float* __restrict__ A,
                    const __bf16* __restrict__ Bpack,
                    const float* __restrict__ bias,
                    float* __restrict__ C,
                    int M)
{
    const int lane  = threadIdx.x & 31;
    const int wave  = threadIdx.x >> 5;
    const int tile  = blockIdx.x * 4 + wave;
    const int ntile = M >> 4;
    if (tile >= ntile) return;                 // wave-uniform exit (EXEC stays full)

    const int row0   = tile * 16;
    const int arow   = row0 + (lane & 15);
    const int khalfA = (lane >> 4) << 3;       // 0 or 8 (A K-half per lane group)

    // A fragments for all 4 K-chunks, per ISA 16-bit A 16x32 layout.
    // Slots 0..7  = K = kc*32 + khalfA + 0..7      (contiguous)
    // Slots 8..15 = K = kc*32 + 16 + khalfA + 0..7 (contiguous)
    const float* ap = A + (size_t)arow * HIDDIM;
    v16bf afrag[4];
#pragma unroll
    for (int kc = 0; kc < 4; ++kc) {
        float4 a0 = *(const float4*)(ap + kc * 32 + khalfA);
        float4 a1 = *(const float4*)(ap + kc * 32 + khalfA + 4);
        float4 a2 = *(const float4*)(ap + kc * 32 + khalfA + 16);
        float4 a3 = *(const float4*)(ap + kc * 32 + khalfA + 20);
        afrag[kc][0]  = f2bf(a0.x); afrag[kc][1]  = f2bf(a0.y);
        afrag[kc][2]  = f2bf(a0.z); afrag[kc][3]  = f2bf(a0.w);
        afrag[kc][4]  = f2bf(a1.x); afrag[kc][5]  = f2bf(a1.y);
        afrag[kc][6]  = f2bf(a1.z); afrag[kc][7]  = f2bf(a1.w);
        afrag[kc][8]  = f2bf(a2.x); afrag[kc][9]  = f2bf(a2.y);
        afrag[kc][10] = f2bf(a2.z); afrag[kc][11] = f2bf(a2.w);
        afrag[kc][12] = f2bf(a3.x); afrag[kc][13] = f2bf(a3.y);
        afrag[kc][14] = f2bf(a3.z); afrag[kc][15] = f2bf(a3.w);
    }

#pragma unroll 1
    for (int nt = 0; nt < HIDDIM / 16; ++nt) {
        const int ncol = nt * 16 + (lane & 15);

        // preload all 4 packed B fragments: v16bf units, index ((nt*4+kc)*32+lane)
        const v16bf* bpl = (const v16bf*)Bpack + (size_t)nt * 4 * 32 + lane;
        v16bf bfrag[4];
#pragma unroll
        for (int kc = 0; kc < 4; ++kc) bfrag[kc] = bpl[kc * 32];

        v8f acc;
        const float bval = bias[ncol];
#pragma unroll
        for (int r = 0; r < 8; ++r) acc[r] = bval;

#pragma unroll
        for (int kc = 0; kc < 4; ++kc) {
            acc = __builtin_amdgcn_wmma_f32_16x16x32_bf16(
                false, afrag[kc], false, bfrag[kc], (short)0, acc, false, false);
        }

        // C/D layout: VGPR r -> M = r + 8*(lane>=16), N = lane&15 ; no guards (M%16==0)
        // fixed stride 128 floats -> r*512B immediate store offsets off one address
        float* cp = C + (size_t)(row0 + ((lane >> 4) << 3)) * HIDDIM + ncol;
#pragma unroll
        for (int r = 0; r < 8; ++r) cp[r * HIDDIM] = acc[r];
    }
}

// ---------------------------------------------------------------------------
// Utility / edge kernels
// ---------------------------------------------------------------------------
__global__ void fill_kernel(float* __restrict__ p, float v, int cnt) {
    int i = blockIdx.x * blockDim.x + threadIdx.x;
    if (i < cnt) p[i] = v;
}

__device__ __forceinline__ void atomicMaxF(float* addr, float val) {
    unsigned int* ia = (unsigned int*)addr;
    unsigned int old = *ia;
    while (__uint_as_float(old) < val) {
        unsigned int assumed = old;
        old = atomicCAS(ia, assumed, __float_as_uint(val));
        if (old == assumed) break;
    }
}

// per-edge attention logits + segment max (CAS-based float max)
template <int HEADS_T, int D_T>
__global__ void edge_logits_kernel(const float* __restrict__ q, const float* __restrict__ k,
                                   const int* __restrict__ src, const int* __restrict__ dst,
                                   float* __restrict__ logits, float* __restrict__ m,
                                   int nedge, float scale)
{
    int e = blockIdx.x * blockDim.x + threadIdx.x;
    if (e >= nedge) return;
    const int s = src[e], t = dst[e];
    const float* qd = q + (size_t)t * (HEADS_T * D_T);
    const float* ks = k + (size_t)s * (HEADS_T * D_T);
#pragma unroll
    for (int h = 0; h < HEADS_T; ++h) {
        float acc = 0.0f;
        for (int j = 0; j < D_T; ++j) acc += qd[h * D_T + j] * ks[h * D_T + j];
        acc *= scale;
        logits[(size_t)e * HEADS_T + h] = acc;
        atomicMaxF(&m[(size_t)t * HEADS_T + h], acc);
    }
}

__global__ void fix_m_kernel(float* __restrict__ m, int cnt) {
    int i = blockIdx.x * blockDim.x + threadIdx.x;
    if (i < cnt) { float v = m[i]; if (!(v > -3.0e38f && v < 3.0e38f)) m[i] = 0.0f; }
}

// ex = exp(logit - m[dst]); denom[dst] += ex   (logits overwritten with ex)
template <int HEADS_T>
__global__ void edge_exp_kernel(float* __restrict__ logits, const float* __restrict__ m,
                                float* __restrict__ denom, const int* __restrict__ dst,
                                int nedge)
{
    int idx = blockIdx.x * blockDim.x + threadIdx.x;
    if (idx >= nedge * HEADS_T) return;
    int e = idx / HEADS_T, h = idx - e * HEADS_T;   // HEADS_T is power of two -> shifts
    int t = dst[e];
    float ex = __expf(logits[idx] - m[(size_t)t * HEADS_T + h]);
    logits[idx] = ex;
    atomicAdd(&denom[(size_t)t * HEADS_T + h], ex);
}

// attn[dst] += scale * (ex/denom) * v[src]
template <int HEADS_T, int D_T>
__global__ void edge_agg_kernel(const float* __restrict__ ex, const float* __restrict__ denom,
                                const float* __restrict__ v,
                                const int* __restrict__ src, const int* __restrict__ dst,
                                float* __restrict__ attn, int nedge, float outScale)
{
    int idx = blockIdx.x * blockDim.x + threadIdx.x;
    if (idx >= nedge * HEADS_T) return;
    int e = idx / HEADS_T, h = idx - e * HEADS_T;
    int s = src[e], t = dst[e];
    float a = ex[idx] / (denom[(size_t)t * HEADS_T + h] + 1e-16f);
    a *= outScale;
    const float* vp = v + (size_t)s * (HEADS_T * D_T) + h * D_T;
    float* op = attn + (size_t)t * (HEADS_T * D_T) + h * D_T;
    for (int j = 0; j < D_T; ++j) atomicAdd(&op[j], a * vp[j]);
}

// beta-gated skip: one wave per node, lane-parallel 3*128 dot + shfl reduce
__global__ void beta_skip_kernel(const float* __restrict__ aout, const float* __restrict__ r,
                                 const float* __restrict__ Wb, float* __restrict__ hout, int n)
{
    int gid  = blockIdx.x * blockDim.x + threadIdx.x;
    int node = gid >> 5;
    int lane = gid & 31;
    if (node >= n) return;
    const float* orow = aout + (size_t)node * HIDDIM;
    const float* rrow = r    + (size_t)node * HIDDIM;
    float acc = 0.0f;
#pragma unroll
    for (int i = 0; i < HIDDIM / 32; ++i) {
        int f = lane + i * 32;
        float o = orow[f], rr = rrow[f];
        acc += o * Wb[f] + rr * Wb[HIDDIM + f] + (o - rr) * Wb[2 * HIDDIM + f];
    }
    for (int off = 16; off > 0; off >>= 1) acc += __shfl_xor(acc, off, 32);
    float g = 1.0f / (1.0f + __expf(-acc));
    float* out = hout + (size_t)node * HIDDIM;
#pragma unroll
    for (int i = 0; i < HIDDIM / 32; ++i) {
        int f = lane + i * 32;
        float o = orow[f], rr = rrow[f];
        out[f] = g * rr + (1.0f - g) * o;
    }
}

// each thread owns one feature column; register-accumulated sum/sumsq
__global__ void bn_stats_kernel(const float* __restrict__ x, int n, float* __restrict__ sums)
{
    int f = threadIdx.x;               // blockDim.x == 128
    float s = 0.0f, q = 0.0f;
    for (int i = blockIdx.x; i < n; i += gridDim.x) {
        float v = x[(size_t)i * HIDDIM + f];
        s += v; q += v * v;
    }
    atomicAdd(&sums[f], s);
    atomicAdd(&sums[HIDDIM + f], q);
}

__global__ void bn_apply_relu_kernel(const float* __restrict__ x, const float* __restrict__ sums,
                                     const float* __restrict__ gamma, const float* __restrict__ beta,
                                     float* __restrict__ y, int n)
{
    int idx = blockIdx.x * blockDim.x + threadIdx.x;
    if (idx >= n * HIDDIM) return;
    int f = idx & (HIDDIM - 1);
    float mu  = sums[f] / (float)n;
    float var = sums[HIDDIM + f] / (float)n - mu * mu;
    float v = (x[idx] - mu) * rsqrtf(var + 1e-5f) * gamma[f] + beta[f];
    y[idx] = v > 0.0f ? v : 0.0f;
}

__global__ void pool_sum_kernel(const float* __restrict__ h, const int* __restrict__ batch,
                                float* __restrict__ psum, int n)
{
    int idx = blockIdx.x * blockDim.x + threadIdx.x;
    if (idx >= n * HIDDIM) return;
    int node = idx >> 7, f = idx & (HIDDIM - 1);
    atomicAdd(&psum[(size_t)batch[node] * HIDDIM + f], h[idx]);
}

__global__ void pool_cnt_kernel(const int* __restrict__ batch, float* __restrict__ pcnt, int n)
{
    int i = blockIdx.x * blockDim.x + threadIdx.x;
    if (i < n) atomicAdd(&pcnt[batch[i]], 1.0f);
}

__global__ void pool_div_kernel(const float* __restrict__ psum, const float* __restrict__ pcnt,
                                float* __restrict__ out)
{
    int idx = blockIdx.x * blockDim.x + threadIdx.x;
    if (idx >= NGRAPH * HIDDIM) return;
    float c = pcnt[idx >> 7];
    out[idx] = psum[idx] / (c > 1.0f ? c : 1.0f);
}

// ---------------------------------------------------------------------------
// Host orchestration
// ---------------------------------------------------------------------------
static inline int divup(int a, int b) { return (a + b - 1) / b; }

extern "C" void kernel_launch(void* const* d_in, const int* in_sizes, int n_in,
                              void* d_out, int out_size, void* d_ws, size_t ws_size,
                              hipStream_t stream)
{
    const float* x     = (const float*)d_in[0];
    const int*   ei    = (const int*)  d_in[1];
    const int*   batch = (const int*)  d_in[2];
    const float* Wp = (const float*)d_in[3];  const float* bp = (const float*)d_in[4];
    const float* Wq = (const float*)d_in[5];  const float* bq = (const float*)d_in[6];
    const float* Wk = (const float*)d_in[7];  const float* bk = (const float*)d_in[8];
    const float* Wv = (const float*)d_in[9];  const float* bv = (const float*)d_in[10];
    const float* Ws = (const float*)d_in[11]; const float* bs = (const float*)d_in[12];
    const float* Wbeta  = (const float*)d_in[13];
    const float* Wq3 = (const float*)d_in[14]; const float* bq3 = (const float*)d_in[15];
    const float* Wk3 = (const float*)d_in[16]; const float* bk3 = (const float*)d_in[17];
    const float* Wv3 = (const float*)d_in[18]; const float* bv3 = (const float*)d_in[19];
    const float* Ws3 = (const float*)d_in[20]; const float* bs3 = (const float*)d_in[21];
    const float* Wbeta3 = (const float*)d_in[22];
    const float* bn_g = (const float*)d_in[23];
    const float* bn_b = (const float*)d_in[24];

    const int* src = ei;
    const int* dst = ei + EDGES;

    // workspace carving (~80 MB)
    char* wsp = (char*)d_ws;
    auto carveB = [&](size_t bytes) -> void* {
        void* p = (void*)wsp;
        wsp += ((bytes + 255) / 256) * 256;
        return p;
    };
    auto carve = [&](size_t nfloat) -> float* { return (float*)carveB(nfloat * 4); };

    float* hA     = carve((size_t)NODES * HIDDIM);
    float* hB     = carve((size_t)NODES * HIDDIM);
    float* qb     = carve((size_t)NODES * HIDDIM);
    float* kb     = carve((size_t)NODES * HIDDIM);
    float* vb     = carve((size_t)NODES * HIDDIM);
    float* rb     = carve((size_t)NODES * HIDDIM);
    float* attn   = carve((size_t)NODES * HIDDIM);
    float* logits = carve((size_t)EDGES * NHEADS);
    float* mbuf   = carve((size_t)NODES * NHEADS);
    float* denom  = carve((size_t)NODES * NHEADS);
    float* bnsums = carve(2 * HIDDIM);
    float* psum   = carve((size_t)NGRAPH * HIDDIM);
    float* pcnt   = carve(NGRAPH);

    const int PACK128 = (HIDDIM / 16) * TILE_ELEMS;          // 16384 bf16
    __bf16* packWp  = (__bf16*)carveB((size_t)PACK128 * 2);
    __bf16* packWq  = (__bf16*)carveB((size_t)3 * PACK128 * 2);
    __bf16* packWk  = (__bf16*)carveB((size_t)3 * PACK128 * 2);
    __bf16* packWv  = (__bf16*)carveB((size_t)3 * PACK128 * 2);
    __bf16* packWs  = (__bf16*)carveB((size_t)3 * PACK128 * 2);
    __bf16* packWq3 = (__bf16*)carveB((size_t)NHEADS * PACK128 * 2);
    __bf16* packWk3 = (__bf16*)carveB((size_t)NHEADS * PACK128 * 2);
    __bf16* packWv3 = (__bf16*)carveB((size_t)NHEADS * PACK128 * 2);
    __bf16* packWs3 = (__bf16*)carveB((size_t)PACK128 * 2);

    auto packB = [&](const float* W, int ldbW, __bf16* dstPack, int Ncols) {
        int total = (Ncols / 16) * TILE_ELEMS;
        pack_b_kernel<<<divup(total, 256), 256, 0, stream>>>(W, ldbW, dstPack, Ncols / 16);
    };
    packB(Wp, HIDDIM, packWp, HIDDIM);
    for (int L = 0; L < 3; ++L) {
        packB(Wq + (size_t)L * HIDDIM * HIDDIM, HIDDIM, packWq + (size_t)L * PACK128, HIDDIM);
        packB(Wk + (size_t)L * HIDDIM * HIDDIM, HIDDIM, packWk + (size_t)L * PACK128, HIDDIM);
        packB(Wv + (size_t)L * HIDDIM * HIDDIM, HIDDIM, packWv + (size_t)L * PACK128, HIDDIM);
        packB(Ws + (size_t)L * HIDDIM * HIDDIM, HIDDIM, packWs + (size_t)L * PACK128, HIDDIM);
    }
    packB(Wq3, NHEADS * HIDDIM, packWq3, NHEADS * HIDDIM);
    packB(Wk3, NHEADS * HIDDIM, packWk3, NHEADS * HIDDIM);
    packB(Wv3, NHEADS * HIDDIM, packWv3, NHEADS * HIDDIM);
    packB(Ws3, HIDDIM, packWs3, HIDDIM);

    const dim3 gblk(128);
    const dim3 ggrd(divup(NODES / 16, 4));
    auto gemm = [&](const float* A, const __bf16* Bp, const float* bias, float* C) {
        wmma_gemm_bias<<<ggrd, gblk, 0, stream>>>(A, Bp, bias, C, NODES);
    };
    auto fill = [&](float* p, float v, int cnt) {
        fill_kernel<<<divup(cnt, 256), 256, 0, stream>>>(p, v, cnt);
    };

    // h = x @ Wp + bp
    gemm(x, packWp, bp, hA);

    // ---- layers 0..2 : heads=8, d=16, concat ----
    for (int L = 0; L < 3; ++L) {
        gemm(hA, packWq + (size_t)L * PACK128, bq + L * HIDDIM, qb);
        gemm(hA, packWk + (size_t)L * PACK128, bk + L * HIDDIM, kb);
        gemm(hA, packWv + (size_t)L * PACK128, bv + L * HIDDIM, vb);
        gemm(hA, packWs + (size_t)L * PACK128, bs + L * HIDDIM, rb);

        fill(mbuf, -3.4e38f, NODES * NHEADS);
        fill(denom, 0.0f, NODES * NHEADS);
        fill(attn, 0.0f, NODES * HIDDIM);

        edge_logits_kernel<NHEADS, 16><<<divup(EDGES, 256), 256, 0, stream>>>(
            qb, kb, src, dst, logits, mbuf, EDGES, 0.25f);
        fix_m_kernel<<<divup(NODES * NHEADS, 256), 256, 0, stream>>>(mbuf, NODES * NHEADS);
        edge_exp_kernel<NHEADS><<<divup(EDGES * NHEADS, 256), 256, 0, stream>>>(
            logits, mbuf, denom, dst, EDGES);
        edge_agg_kernel<NHEADS, 16><<<divup(EDGES * NHEADS, 256), 256, 0, stream>>>(
            logits, denom, vb, src, dst, attn, EDGES, 1.0f);

        beta_skip_kernel<<<divup(NODES * 32, 128), 128, 0, stream>>>(
            attn, rb, Wbeta + (size_t)L * 3 * HIDDIM, hB, NODES);

        fill(bnsums, 0.0f, 2 * HIDDIM);
        bn_stats_kernel<<<240, HIDDIM, 0, stream>>>(hB, NODES, bnsums);
        bn_apply_relu_kernel<<<divup(NODES * HIDDIM, 256), 256, 0, stream>>>(
            hB, bnsums, bn_g + L * HIDDIM, bn_b + L * HIDDIM, hA, NODES);
    }

    // ---- layer 3 : heads=8, d=128, head-mean; processed one head at a time ----
    fill(attn, 0.0f, NODES * HIDDIM);
    const float scale3 = 1.0f / sqrtf(128.0f);
    for (int h = 0; h < NHEADS; ++h) {
        // per-head column slice of the packed weight = contiguous PACK128 block
        gemm(hA, packWq3 + (size_t)h * PACK128, bq3 + h * HIDDIM, qb);
        gemm(hA, packWk3 + (size_t)h * PACK128, bk3 + h * HIDDIM, kb);
        gemm(hA, packWv3 + (size_t)h * PACK128, bv3 + h * HIDDIM, vb);

        fill(mbuf, -3.4e38f, NODES);
        fill(denom, 0.0f, NODES);
        edge_logits_kernel<1, HIDDIM><<<divup(EDGES, 256), 256, 0, stream>>>(
            qb, kb, src, dst, logits, mbuf, EDGES, scale3);
        fix_m_kernel<<<divup(NODES, 256), 256, 0, stream>>>(mbuf, NODES);
        edge_exp_kernel<1><<<divup(EDGES, 256), 256, 0, stream>>>(
            logits, mbuf, denom, dst, EDGES);
        edge_agg_kernel<1, HIDDIM><<<divup(EDGES, 256), 256, 0, stream>>>(
            logits, denom, vb, src, dst, attn, EDGES, 1.0f / NHEADS);
    }
    gemm(hA, packWs3, bs3, rb);
    beta_skip_kernel<<<divup(NODES * 32, 128), 128, 0, stream>>>(attn, rb, Wbeta3, hB, NODES);

    fill(bnsums, 0.0f, 2 * HIDDIM);
    bn_stats_kernel<<<240, HIDDIM, 0, stream>>>(hB, NODES, bnsums);
    bn_apply_relu_kernel<<<divup(NODES * HIDDIM, 256), 256, 0, stream>>>(
        hB, bnsums, bn_g + 3 * HIDDIM, bn_b + 3 * HIDDIM, hA, NODES);

    // ---- global mean pool per graph ----
    fill(psum, 0.0f, NGRAPH * HIDDIM);
    fill(pcnt, 0.0f, NGRAPH);
    pool_sum_kernel<<<divup(NODES * HIDDIM, 256), 256, 0, stream>>>(hA, batch, psum, NODES);
    pool_cnt_kernel<<<divup(NODES, 256), 256, 0, stream>>>(batch, pcnt, NODES);
    pool_div_kernel<<<divup(NGRAPH * HIDDIM, 256), 256, 0, stream>>>(psum, pcnt, (float*)d_out);
}
